// MultiTaskModel_44100724196048
// MI455X (gfx1250) — matrix operand order repaired and checked by reading, not compiled, not run on hardware
//
#include <hip/hip_runtime.h>
#include <hip/hip_bf16.h>

typedef __attribute__((ext_vector_type(16))) __bf16 v16bf;
typedef __attribute__((ext_vector_type(8)))  float  v8f;

#define EMBED_DIM   64
#define BATCH       1024
#define NUM_CAND    200000
#define TOPK        10
#define NSPLIT      32
#define NTILES      (NUM_CAND / 16)                        // 12500
#define TILES_PER_SPLIT ((NTILES + NSPLIT - 1) / NSPLIT)   // 391
#define NLISTS      (2 * NSPLIT)                           // partial lists per user
#define NEG_BIG     (-3.0e38f)

// ---------------------------------------------------------------------------
// sorted-descending insertion (merge kernel only; small, runs once per user)
// tie-break: lower candidate index wins (jax.lax.top_k stability)
// ---------------------------------------------------------------------------
__device__ __forceinline__ void topk_insert(float (&vals)[TOPK], int (&inds)[TOPK],
                                            float v, int ci) {
  if (v > vals[TOPK - 1] || (v == vals[TOPK - 1] && ci < inds[TOPK - 1])) {
    vals[TOPK - 1] = v;
    inds[TOPK - 1] = ci;
#pragma unroll
    for (int q = TOPK - 1; q > 0; --q) {
      bool up = (vals[q] > vals[q - 1]) ||
                (vals[q] == vals[q - 1] && inds[q] < inds[q - 1]);
      if (up) {
        float tv = vals[q]; vals[q] = vals[q - 1]; vals[q - 1] = tv;
        int   ti = inds[q]; inds[q] = inds[q - 1]; inds[q - 1] = ti;
      }
    }
  }
}

__device__ __forceinline__ v16bf cvt16(float4 q0, float4 q1, float4 q2, float4 q3) {
  v16bf b;
  b[0]  = (__bf16)q0.x; b[1]  = (__bf16)q0.y; b[2]  = (__bf16)q0.z; b[3]  = (__bf16)q0.w;
  b[4]  = (__bf16)q1.x; b[5]  = (__bf16)q1.y; b[6]  = (__bf16)q1.z; b[7]  = (__bf16)q1.w;
  b[8]  = (__bf16)q2.x; b[9]  = (__bf16)q2.y; b[10] = (__bf16)q2.z; b[11] = (__bf16)q2.w;
  b[12] = (__bf16)q3.x; b[13] = (__bf16)q3.y; b[14] = (__bf16)q3.z; b[15] = (__bf16)q3.w;
  return b;
}

// ---------------------------------------------------------------------------
// Fused scores GEMM (bf16 WMMA, f32 accum) + per-user top-10 with a
// wave-uniform threshold guard so untouched tiles cost ~zero beyond the GEMM.
// grid = (64 user tiles, NSPLIT candidate splits); block = 32 (one wave32).
// Lane L owns the top-10 list of (user = L&15, candidate-half = L>>4).
// ---------------------------------------------------------------------------
__global__ __launch_bounds__(32) void scores_topk_kernel(
    const int*   __restrict__ user_id,
    const float* __restrict__ user_table,
    const float* __restrict__ cand_table,
    float*       __restrict__ part_val,
    int*         __restrict__ part_idx) {
  __shared__ float lds_u[16 * EMBED_DIM];   // staged user embeddings (f32)
  __shared__ float lds_s[16 * 17];          // 16x16 score tile (padded)
  __shared__ float lds_th[32];              // th[half*16 + user] = owner's current min

  const int lane = threadIdx.x;
  const int ut   = blockIdx.x;
  const int sp   = blockIdx.y;

  // Stage 16 user-embedding rows (gather) into LDS.
  for (int i = lane; i < 16 * EMBED_DIM; i += 32) {
    int r = i >> 6;
    int k = i & 63;
    int uid = user_id[ut * 16 + r];
    lds_u[i] = user_table[(size_t)uid * EMBED_DIM + k];
  }
  lds_th[lane] = NEG_BIG;
  __syncthreads();

  const int m = lane & 15;
  const int h = lane >> 4;

  // A fragments (16x32 bf16 each, contiguous k = 16h+e, as B), invariant over candidate loop.
  v16bf a0, a1;
#pragma unroll
  for (int e = 0; e < 16; ++e) {
    int kb = 16 * h + e;
    a0[e] = (__bf16)lds_u[m * EMBED_DIM + kb];
    a1[e] = (__bf16)lds_u[m * EMBED_DIM + 32 + kb];
  }

  // Owner state: unsorted top-10 + tracked min slot.
  float tv[TOPK]; int tx[TOPK];
#pragma unroll
  for (int q = 0; q < TOPK; ++q) { tv[q] = NEG_BIG; tx[q] = 0x7fffffff; }
  float mn  = NEG_BIG;
  int   mni = 0x7fffffff;
  int   mnp = 0;

  // Guard thresholds for the 8 users this lane's accumulators cover, at the
  // candidate-half hp = (column m >= 8). Refreshed only after triggered tiles.
  const int hp = (m >= 8) ? 1 : 0;
  float thr[8];
#pragma unroll
  for (int r = 0; r < 8; ++r) thr[r] = NEG_BIG;

  int t0 = sp * TILES_PER_SPLIT;
  int t1 = t0 + TILES_PER_SPLIT;
  if (t1 > NTILES) t1 = NTILES;

  for (int t = t0; t < t1; ++t) {
    const int c0 = t * 16;
    const float4* p4 = (const float4*)(cand_table + (size_t)(c0 + m) * EMBED_DIM);
    if (t + 1 < t1) {
      __builtin_prefetch((const void*)(cand_table + (size_t)(c0 + 16 + m) * EMBED_DIM), 0, 0);
    }
    float4 r0 = p4[4 * h + 0], r1 = p4[4 * h + 1], r2 = p4[4 * h + 2], r3 = p4[4 * h + 3];
    float4 s0 = p4[8 + 4 * h + 0], s1 = p4[8 + 4 * h + 1],
           s2 = p4[8 + 4 * h + 2], s3 = p4[8 + 4 * h + 3];
    v16bf b0 = cvt16(r0, r1, r2, r3);   // K = 0..31
    v16bf b1 = cvt16(s0, s1, s2, s3);   // K = 32..63

    v8f acc = {0.f, 0.f, 0.f, 0.f, 0.f, 0.f, 0.f, 0.f};
    acc = __builtin_amdgcn_wmma_f32_16x16x32_bf16(false, a0, false, b0,
                                                  (short)0, acc, false, false);
    acc = __builtin_amdgcn_wmma_f32_16x16x32_bf16(false, a1, false, b1,
                                                  (short)0, acc, false, false);

    // Wave-uniform guard: can any score enter its owner's list?
    bool hit = false;
#pragma unroll
    for (int r = 0; r < 8; ++r) hit = hit || (acc[r] >= thr[r]);

    if (__ballot(hit)) {
      // Transpose tile through LDS; owner lanes scan their 8 candidates.
#pragma unroll
      for (int r = 0; r < 8; ++r) lds_s[(8 * h + r) * 17 + m] = acc[r];
      __syncthreads();
#pragma unroll
      for (int j = 0; j < 8; ++j) {
        float v  = lds_s[m * 17 + 8 * h + j];
        int   ci = c0 + 8 * h + j;
        if (v > mn || (v == mn && ci < mni)) {
          // replace current min slot
#pragma unroll
          for (int q = 0; q < TOPK; ++q)
            if (q == mnp) { tv[q] = v; tx[q] = ci; }
          // recompute min (straight-line, predicated)
          mn = tv[0]; mni = tx[0]; mnp = 0;
#pragma unroll
          for (int q = 1; q < TOPK; ++q) {
            bool lt = (tv[q] < mn) || (tv[q] == mn && tx[q] > mni);
            if (lt) { mn = tv[q]; mni = tx[q]; mnp = q; }
          }
        }
      }
      // Publish new thresholds, then refresh guard registers.
      lds_th[h * 16 + m] = mn;
      __syncthreads();
      float4 ta = *(const float4*)&lds_th[hp * 16 + 8 * h];
      float4 tb = *(const float4*)&lds_th[hp * 16 + 8 * h + 4];
      thr[0] = ta.x; thr[1] = ta.y; thr[2] = ta.z; thr[3] = ta.w;
      thr[4] = tb.x; thr[5] = tb.y; thr[6] = tb.z; thr[7] = tb.w;
      __syncthreads();
    }
  }

  // Emit this lane's (unsorted) partial list.
  const int u    = ut * 16 + m;
  const int list = sp * 2 + h;
  const int base = (u * NLISTS + list) * TOPK;
#pragma unroll
  for (int q = 0; q < TOPK; ++q) {
    part_val[base + q] = tv[q];
    part_idx[base + q] = tx[q];
  }
}

// ---------------------------------------------------------------------------
// Final merge: NLISTS unsorted partial lists per user -> sorted top-10 (f32 idx).
// ---------------------------------------------------------------------------
__global__ void topk_merge_kernel(const float* __restrict__ part_val,
                                  const int*   __restrict__ part_idx,
                                  float*       __restrict__ out_pred) {
  int u = blockIdx.x * blockDim.x + threadIdx.x;
  if (u >= BATCH) return;
  float fv[TOPK]; int fi[TOPK];
#pragma unroll
  for (int q = 0; q < TOPK; ++q) { fv[q] = NEG_BIG; fi[q] = 0x7fffffff; }
  for (int s = 0; s < NLISTS; ++s) {
    int base = (u * NLISTS + s) * TOPK;
    for (int q = 0; q < TOPK; ++q) {
      topk_insert(fv, fi, part_val[base + q], part_idx[base + q]);
    }
  }
  for (int q = 0; q < TOPK; ++q) out_pred[u * TOPK + q] = (float)fi[q];
}

// ---------------------------------------------------------------------------
// Embedding gather: user/cand embedding outputs + MLP input h=[ue|ce].
// ---------------------------------------------------------------------------
__global__ void gather_embed_kernel(const int* __restrict__ user_id,
                                    const int* __restrict__ movie_id,
                                    const float* __restrict__ user_table,
                                    const float* __restrict__ cand_table,
                                    float* __restrict__ out_ue,
                                    float* __restrict__ out_ce,
                                    float* __restrict__ h) {
  int i = blockIdx.x * blockDim.x + threadIdx.x;
  int b = i >> 6;
  int d = i & 63;
  float ue = user_table[(size_t)user_id[b] * EMBED_DIM + d];
  float ce = cand_table[(size_t)movie_id[b] * EMBED_DIM + d];
  out_ue[i] = ue;
  out_ce[i] = ce;
  h[b * 128 + d]      = ue;
  h[b * 128 + 64 + d] = ce;
}

// ---------------------------------------------------------------------------
// WMMA dense layer: y = act(x @ W + b), x:[1024,K] f32, W:[K,N] f32 row-major.
// bf16 inputs, f32 accumulation; bias+relu epilogue in f32.
// grid = (BATCH/16, N/16); block = 32 (one wave32).
// ---------------------------------------------------------------------------
template <int K, int N, int RELU>
__global__ __launch_bounds__(32) void fc_wmma_kernel(
    const float* __restrict__ x, const float* __restrict__ W,
    const float* __restrict__ bias, float* __restrict__ y) {
  __shared__ float lds_x[16 * K];
  const int lane = threadIdx.x;
  const int ut   = blockIdx.x;
  const int nt   = blockIdx.y;

  for (int i = lane; i < 16 * K; i += 32)
    lds_x[i] = x[(size_t)(ut * 16 + i / K) * K + (i % K)];
  __syncthreads();

  const int m   = lane & 15;
  const int h   = lane >> 4;
  const int col = nt * 16 + m;

  v8f acc = {0.f, 0.f, 0.f, 0.f, 0.f, 0.f, 0.f, 0.f};
#pragma unroll
  for (int kc = 0; kc < K; kc += 32) {
    v16bf a, b;
#pragma unroll
    for (int e = 0; e < 16; ++e) {
      int kb = (e < 8) ? (8 * h + e) : (16 + 8 * h + (e - 8));
      a[e] = (__bf16)lds_x[m * K + kc + kb];
      b[e] = (__bf16)W[(size_t)(kc + kb) * N + col];
    }
    acc = __builtin_amdgcn_wmma_f32_16x16x32_bf16(false, a, false, b,
                                                  (short)0, acc, false, false);
  }

  float bc = bias[col];
#pragma unroll
  for (int r = 0; r < 8; ++r) {
    int   row = ut * 16 + 8 * h + r;
    float v   = acc[r] + bc;
    if (RELU) v = fmaxf(v, 0.f);
    y[(size_t)row * N + col] = v;
  }
}

// ---------------------------------------------------------------------------
// Scalar dense layer (final N=1 projection).
// ---------------------------------------------------------------------------
__global__ void fc_kernel(const float* __restrict__ x, const float* __restrict__ W,
                          const float* __restrict__ bias, float* __restrict__ y,
                          int K, int N, int relu) {
  int i = blockIdx.x * blockDim.x + threadIdx.x;
  int b = i / N;
  int j = i - b * N;
  if (b >= BATCH) return;
  float acc = bias[j];
  const float* xr = x + b * K;
  for (int k = 0; k < K; ++k) acc = fmaf(xr[k], W[k * N + j], acc);
  if (relu) acc = fmaxf(acc, 0.f);
  y[i] = acc;
}

// ---------------------------------------------------------------------------
extern "C" void kernel_launch(void* const* d_in, const int* in_sizes, int n_in,
                              void* d_out, int out_size, void* d_ws, size_t ws_size,
                              hipStream_t stream) {
  (void)in_sizes; (void)n_in; (void)out_size; (void)ws_size;
  const int*   user_id    = (const int*)d_in[0];
  const int*   movie_id   = (const int*)d_in[1];
  const float* user_table = (const float*)d_in[2];
  const float* cand_table = (const float*)d_in[3];
  const float* W1 = (const float*)d_in[4];
  const float* b1 = (const float*)d_in[5];
  const float* W2 = (const float*)d_in[6];
  const float* b2 = (const float*)d_in[7];
  const float* W3 = (const float*)d_in[8];
  const float* b3 = (const float*)d_in[9];

  float* out    = (float*)d_out;
  float* out_ue = out;                                   // [1024,64]
  float* out_ce = out + BATCH * EMBED_DIM;               // [1024,64]
  float* out_rt = out + 2 * BATCH * EMBED_DIM;           // [1024,1]
  float* out_pr = out + 2 * BATCH * EMBED_DIM + BATCH;   // [1024,10]

  char*  w        = (char*)d_ws;
  float* part_val = (float*)(w + 0);          // 1024*64*10 f = 2,621,440 B
  int*   part_idx = (int*)  (w + 2621440);    // 2,621,440 B
  float* h        = (float*)(w + 5242880);    //   524,288 B  [1024,128]
  float* h1       = (float*)(w + 5767168);    // 1,048,576 B  [1024,256]
  float* h2       = (float*)(w + 6815744);    //   524,288 B  [1024,128]

  // Embedding gathers (+ MLP input build)
  gather_embed_kernel<<<(BATCH * EMBED_DIM) / 256, 256, 0, stream>>>(
      user_id, movie_id, user_table, cand_table, out_ue, out_ce, h);

  // Rating MLP: 128 -> 256 (relu) -> 128 (relu) -> 1
  fc_wmma_kernel<128, 256, 1><<<dim3(BATCH / 16, 256 / 16), 32, 0, stream>>>(h,  W1, b1, h1);
  fc_wmma_kernel<256, 128, 1><<<dim3(BATCH / 16, 128 / 16), 32, 0, stream>>>(h1, W2, b2, h2);
  fc_kernel<<<(BATCH + 255) / 256, 256, 0, stream>>>(h2, W3, b3, out_rt, 128, 1, 0);

  // Fused scores GEMM (bf16 WMMA) + top-k
  scores_topk_kernel<<<dim3(BATCH / 16, NSPLIT), 32, 0, stream>>>(
      user_id, user_table, cand_table, part_val, part_idx);
  topk_merge_kernel<<<(BATCH + 255) / 256, 256, 0, stream>>>(
      part_val, part_idx, out_pr);
}
